// MultiHeadAttention_36464272343255
// MI455X (gfx1250) — matrix operand order repaired and checked by reading, not compiled, not run on hardware
//
#include <hip/hip_runtime.h>
#include <hip/hip_bf16.h>
#include <math.h>

#define B_ 2
#define S_ 2048
#define E_ 1024
#define H_ 16
#define DC_ 256
#define DR_ 128
#define HD_ 64
#define M_ (B_*S_)      /* 4096 tokens */
#define NC_ (E_-DR_)    /* 896 */
#define QT2_ (S_/32)    /* 64 32-query tiles per (b,h) */

typedef _Float16 half_t;
typedef __attribute__((ext_vector_type(16))) _Float16 v16h;
typedef __attribute__((ext_vector_type(8)))  _Float16 v8h;
typedef __attribute__((ext_vector_type(8)))  float    v8f;

// ---------------------------------------------------------------------------
// WMMA helpers (CDNA5 wave32; V_WMMA_F32_16X16X32_F16)
// ---------------------------------------------------------------------------
__device__ __forceinline__ v8f wmma_f16(v16h a, v16h b, v8f c) {
  // 8 args: (neg_a, A, neg_b, B, c_mod, C, reuse_a, reuse_b)
  return __builtin_amdgcn_wmma_f32_16x16x32_f16(false, a, false, b, (short)0, c,
                                                false, false);
}

__device__ __forceinline__ float gelu_f(float x) {
  return 0.5f * x * (1.0f + erff(x * 0.70710678118654752f));
}

// A fragment 16x32 (MxK), row-major A with leading dim ld.
// Contiguous k order (as in load_frag_b): lane L: row = L%16; elem e -> K = (L/16)*16 + e  (two contiguous 16B chunks)
__device__ __forceinline__ v16h load_frag_a(const half_t* __restrict__ base,
                                            int ld, int lane) {
  int r = lane & 15, g = lane >> 4;
  const half_t* p = base + (size_t)r * ld + g * 16;
  v8h lo = *(const v8h*)p;
  v8h hi = *(const v8h*)(p + 8);
  v16h out;
#pragma unroll
  for (int i = 0; i < 8; ++i) { out[i] = lo[i]; out[i + 8] = hi[i]; }
  return out;
}

// B fragment 32x16 (KxN) taken from an N x K row-major (i.e. pre-transposed)
// matrix: lane L: col n = L%16; elem e -> K = (L/16)*16 + e  (one 32B chunk)
__device__ __forceinline__ v16h load_frag_b(const half_t* __restrict__ base,
                                            int ld, int lane) {
  int n = lane & 15, g = lane >> 4;
  return *(const v16h*)(base + (size_t)n * ld + g * 16);
}

// ---------------------------------------------------------------------------
// Generic WMMA GEMM: C[M x N] = act(A[M x K] * Wt[N x K]^T + bias)
// grid = (M/64, N/128), block = 128 (4 waves, each a 32x64 tile: 8 wmma/K-step)
// ---------------------------------------------------------------------------
template<bool GELU, typename OUT>
__global__ __launch_bounds__(128) void gemm_wmma_kernel(
    const half_t* __restrict__ A, const half_t* __restrict__ Wt,
    const float* __restrict__ bias, OUT* __restrict__ C,
    int K, int ldc) {
  int lane = threadIdx.x & 31;
  int wave = threadIdx.x >> 5;
  int m0 = blockIdx.x * 64 + (wave >> 1) * 32;
  int n0 = blockIdx.y * 128 + (wave & 1) * 64;

  v8f acc[2][4] = {};
  for (int k0 = 0; k0 < K; k0 += 32) {
    v16h a[2];
    a[0] = load_frag_a(A + (size_t)m0 * K + k0, K, lane);
    a[1] = load_frag_a(A + (size_t)(m0 + 16) * K + k0, K, lane);
    v16h b[4];
#pragma unroll
    for (int ni = 0; ni < 4; ++ni)
      b[ni] = load_frag_b(Wt + (size_t)(n0 + ni * 16) * K + k0, K, lane);

    if (k0 + 32 < K) {  // prefetch next K-slab (global_prefetch_b8)
      __builtin_prefetch(A + (size_t)(m0 + lane) * K + k0 + 32, 0, 3);
      __builtin_prefetch(Wt + (size_t)(n0 + lane) * K + k0 + 32, 0, 3);
      __builtin_prefetch(Wt + (size_t)(n0 + 32 + lane) * K + k0 + 32, 0, 3);
    }
#pragma unroll
    for (int mi = 0; mi < 2; ++mi)
#pragma unroll
      for (int ni = 0; ni < 4; ++ni)
        acc[mi][ni] = wmma_f16(a[mi], b[ni], acc[mi][ni]);
  }

  int cn = lane & 15, g = lane >> 4;
#pragma unroll
  for (int mi = 0; mi < 2; ++mi)
#pragma unroll
    for (int ni = 0; ni < 4; ++ni)
#pragma unroll
      for (int r = 0; r < 8; ++r) {
        int row = m0 + mi * 16 + g * 8 + r;
        int col = n0 + ni * 16 + cn;
        float v = acc[mi][ni][r] + bias[col];
        if (GELU) v = gelu_f(v);
        C[(size_t)row * ldc + col] = (OUT)v;
      }
}

// ---------------------------------------------------------------------------
// Pointwise kernels
// ---------------------------------------------------------------------------
__global__ void f32_to_f16_kernel(const float* __restrict__ in,
                                  half_t* __restrict__ out, int n) {
  int i = blockIdx.x * blockDim.x + threadIdx.x;
  if (i < n) out[i] = (half_t)in[i];
}

// w[din x dout] fp32 -> wt[dout x din] f16
__global__ void transpose_f16_kernel(const float* __restrict__ w,
                                     half_t* __restrict__ wt,
                                     int din, int dout) {
  int i = blockIdx.x * blockDim.x + threadIdx.x;
  if (i >= din * dout) return;
  int r = i / dout, c = i - r * dout;
  wt[(size_t)c * din + r] = (half_t)w[i];
}

// full[M_ x E_] fp32 (concat(c-part, rope-part)) ->
// rope on cols [896,1024), gelu all, permute to (b,h,s,d) f16
__global__ void rope_gelu_perm_kernel(const float* __restrict__ full,
                                      half_t* __restrict__ outh) {
  int i = blockIdx.x * blockDim.x + threadIdx.x;
  if (i >= M_ * E_) return;
  int c = i & (E_ - 1);
  int tok = i >> 10;          // /E_
  int s = tok & (S_ - 1);
  int b = tok >> 11;          // /S_
  float v = full[i];
  if (c >= NC_) {
    int j = c - NC_;          // 0..127
    int fi = j & 63;          // frequency index
    // inv_freq = 10000^(-fi/64) = exp(-fi * ln(10000)/64)
    float ang = (float)s * __expf(-(float)fi * 0.14391565505088738f);
    float cs = cosf(ang), sn = sinf(ang);
    float partner = (j < 64) ? -full[i + 64] : full[i - 64];
    v = v * cs + partner * sn;
  }
  v = gelu_f(v);
  int h = c >> 6, d = c & 63;
  outh[(((size_t)b * H_ + h) * S_ + s) * HD_ + d] = (half_t)v;
}

// full[M_ x E_] fp32 (v) -> vt (b,h,d,s) f16 so P*V B-fragments are contiguous
__global__ void v_perm_kernel(const float* __restrict__ full,
                              half_t* __restrict__ vt) {
  int i = blockIdx.x * blockDim.x + threadIdx.x;
  if (i >= M_ * E_) return;
  int c = i & (E_ - 1);
  int tok = i >> 10;
  int s = tok & (S_ - 1);
  int b = tok >> 11;
  int h = c >> 6, d = c & 63;
  vt[(((size_t)b * H_ + h) * HD_ + d) * S_ + s] = (half_t)full[i];
}

// ---------------------------------------------------------------------------
// Flash attention: one wave per (b,h,32-query tile); key blocks of 32.
// Two 16-query sub-tiles share every K/V fragment (2x L2 traffic reduction).
// qh/kh: (b,h,s,d) f16;  vt: (b,h,d,s) f16;  ao: (b,s,E) f16
// ---------------------------------------------------------------------------
__global__ __launch_bounds__(32) void attn_kernel(
    const half_t* __restrict__ qh, const half_t* __restrict__ kh,
    const half_t* __restrict__ vt, half_t* __restrict__ ao) {
  __shared__ __align__(16) half_t pbuf[2 * 16 * 32];

  int lane = threadIdx.x & 31;
  int g = lane >> 4, cn = lane & 15;
  int bid = blockIdx.x;
  int qt = bid % QT2_;          // 32-query tile index
  int bh = bid / QT2_;          // b*H_ + h
  int h = bh % H_;
  int b = bh / H_;

  const half_t* qbase = qh + (size_t)bh * S_ * HD_;
  const half_t* kbase = kh + (size_t)bh * S_ * HD_;
  const half_t* vbase = vt + (size_t)bh * HD_ * S_;

  // Q A-fragments for both 16-query sub-tiles, K-dims [0,32) and [32,64)
  v16h qa[2][2];
#pragma unroll
  for (int t = 0; t < 2; ++t) {
    qa[t][0] = load_frag_a(qbase + (size_t)(qt * 32 + t * 16) * HD_ + 0, HD_, lane);
    qa[t][1] = load_frag_a(qbase + (size_t)(qt * 32 + t * 16) * HD_ + 32, HD_, lane);
  }

  v8f o[2][4] = {};
  float mi[2][8], li[2][8];
#pragma unroll
  for (int t = 0; t < 2; ++t)
#pragma unroll
    for (int r = 0; r < 8; ++r) { mi[t][r] = -__builtin_inff(); li[t][r] = 0.0f; }

  const float scale = 0.125f;   // 1/sqrt(64)
  int nkb = qt + 1;             // causal: keys [0, 32*(qt+1))

  for (int kb = 0; kb < nkb; ++kb) {
    int kcol0 = kb * 32;

    // Shared K fragments for this 32-key block
    v16h kf[2][2];
#pragma unroll
    for (int hlf = 0; hlf < 2; ++hlf) {
      kf[hlf][0] = load_frag_b(kbase + (size_t)(kcol0 + hlf * 16) * HD_ + 0, HD_, lane);
      kf[hlf][1] = load_frag_b(kbase + (size_t)(kcol0 + hlf * 16) * HD_ + 32, HD_, lane);
    }
    if (kb + 1 < nkb) {  // prefetch next key block (K rows + V rows)
      __builtin_prefetch(kbase + (size_t)(kcol0 + 32 + lane) * HD_, 0, 3);
      __builtin_prefetch(vbase + (size_t)lane * S_ + kcol0 + 32, 0, 3);
      __builtin_prefetch(vbase + (size_t)(lane + 32) * S_ + kcol0 + 32, 0, 3);
    }

    // Scores for both sub-tiles: s[t][hlf] = Q_t * K_half^T
    v8f s01[2][2];
#pragma unroll
    for (int t = 0; t < 2; ++t)
#pragma unroll
      for (int hlf = 0; hlf < 2; ++hlf) {
        v8f c = {};
        c = wmma_f16(qa[t][0], kf[hlf][0], c);
        c = wmma_f16(qa[t][1], kf[hlf][1], c);
        s01[t][hlf] = c;
      }

    // Online softmax per sub-tile
#pragma unroll
    for (int t = 0; t < 2; ++t) {
      float rowmax[8];
#pragma unroll
      for (int r = 0; r < 8; ++r) rowmax[r] = -__builtin_inff();
#pragma unroll
      for (int hlf = 0; hlf < 2; ++hlf) {
        int kcol = kcol0 + hlf * 16 + cn;
#pragma unroll
        for (int r = 0; r < 8; ++r) {
          int row = qt * 32 + t * 16 + g * 8 + r;
          float v = s01[t][hlf][r] * scale;
          if (kcol > row) v = -__builtin_inff();
          s01[t][hlf][r] = v;
          rowmax[r] = fmaxf(rowmax[r], v);
        }
      }
#pragma unroll
      for (int off = 1; off < 16; off <<= 1)
#pragma unroll
        for (int r = 0; r < 8; ++r)
          rowmax[r] = fmaxf(rowmax[r], __shfl_xor(rowmax[r], off, 32));

      float alpha[8];
#pragma unroll
      for (int r = 0; r < 8; ++r) {
        float mnew = fmaxf(mi[t][r], rowmax[r]);
        alpha[r] = __expf(mi[t][r] - mnew);
        mi[t][r] = mnew;
      }

      float rowsum[8];
#pragma unroll
      for (int r = 0; r < 8; ++r) rowsum[r] = 0.0f;
#pragma unroll
      for (int hlf = 0; hlf < 2; ++hlf)
#pragma unroll
        for (int r = 0; r < 8; ++r) {
          float p = __expf(s01[t][hlf][r] - mi[t][r]);
          s01[t][hlf][r] = p;
          rowsum[r] += p;
        }
#pragma unroll
      for (int off = 1; off < 16; off <<= 1)
#pragma unroll
        for (int r = 0; r < 8; ++r)
          rowsum[r] += __shfl_xor(rowsum[r], off, 32);
#pragma unroll
      for (int r = 0; r < 8; ++r) li[t][r] = li[t][r] * alpha[r] + rowsum[r];
#pragma unroll
      for (int f = 0; f < 4; ++f)
#pragma unroll
        for (int r = 0; r < 8; ++r) o[t][f][r] *= alpha[r];
    }

    // C-layout -> A-layout for P via LDS (2 x 16 x 32 f16)
    __syncthreads();
#pragma unroll
    for (int t = 0; t < 2; ++t)
#pragma unroll
      for (int hlf = 0; hlf < 2; ++hlf)
#pragma unroll
        for (int r = 0; r < 8; ++r)
          pbuf[t * 512 + (g * 8 + r) * 32 + hlf * 16 + cn] = (half_t)s01[t][hlf][r];
    __syncthreads();

    v16h pa[2];
#pragma unroll
    for (int t = 0; t < 2; ++t) {
      const half_t* p = &pbuf[t * 512 + (lane & 15) * 32 + g * 16];
      v8h lo = *(const v8h*)p;
      v8h hi = *(const v8h*)(p + 8);
#pragma unroll
      for (int i2 = 0; i2 < 8; ++i2) { pa[t][i2] = lo[i2]; pa[t][i2 + 8] = hi[i2]; }
    }

    // Shared V fragments; O_t += P_t * V
#pragma unroll
    for (int f = 0; f < 4; ++f) {
      v16h vb = *(const v16h*)(vbase + (size_t)(f * 16 + cn) * S_ + kcol0 + g * 16);
      o[0][f] = wmma_f16(pa[0], vb, o[0][f]);
      o[1][f] = wmma_f16(pa[1], vb, o[1][f]);
    }
  }

  // finalize: O / l, write to (b,s,E) f16
#pragma unroll
  for (int t = 0; t < 2; ++t)
#pragma unroll
    for (int r = 0; r < 8; ++r) {
      int tok = qt * 32 + t * 16 + g * 8 + r;
      float inv = 1.0f / li[t][r];
      size_t base = ((size_t)b * S_ + tok) * E_ + h * HD_ + cn;
      ao[base + 0]  = (half_t)(o[t][0][r] * inv);
      ao[base + 16] = (half_t)(o[t][1][r] * inv);
      ao[base + 32] = (half_t)(o[t][2][r] * inv);
      ao[base + 48] = (half_t)(o[t][3][r] * inv);
    }
}

// ---------------------------------------------------------------------------
// Host orchestration
// ---------------------------------------------------------------------------
extern "C" void kernel_launch(void* const* d_in, const int* in_sizes, int n_in,
                              void* d_out, int out_size, void* d_ws, size_t ws_size,
                              hipStream_t stream) {
  (void)in_sizes; (void)n_in; (void)out_size; (void)ws_size;
  const float* x   = (const float*)d_in[0];
  const float* wq  = (const float*)d_in[1];
  const float* bq  = (const float*)d_in[2];
  const float* wqc = (const float*)d_in[3];
  const float* bqc = (const float*)d_in[4];
  const float* wqr = (const float*)d_in[5];
  const float* bqr = (const float*)d_in[6];
  const float* wkv = (const float*)d_in[7];
  const float* bkv = (const float*)d_in[8];
  const float* wkr = (const float*)d_in[9];
  const float* bkr = (const float*)d_in[10];
  const float* wkc = (const float*)d_in[11];
  const float* bkc = (const float*)d_in[12];
  const float* wv  = (const float*)d_in[13];
  const float* bv  = (const float*)d_in[14];
  const float* wo  = (const float*)d_in[15];
  const float* bo  = (const float*)d_in[16];
  float* out = (float*)d_out;

  char* ws = (char*)d_ws;
  size_t off = 0;
  auto alloc = [&](size_t bytes) {
    char* p = ws + off;
    off += (bytes + 255) & ~(size_t)255;
    return p;
  };
  half_t* xh    = (half_t*)alloc((size_t)M_ * E_ * 2);     // x in f16
  half_t* cqh   = (half_t*)alloc((size_t)M_ * DC_ * 2);    // gelu(x wq + bq)
  half_t* ckvh  = (half_t*)alloc((size_t)M_ * DC_ * 2);
  half_t* qhh   = (half_t*)alloc((size_t)M_ * E_ * 2);     // q (b,h,s,d)
  half_t* khh   = (half_t*)alloc((size_t)M_ * E_ * 2);     // k (b,h,s,d)
  half_t* vth   = (half_t*)alloc((size_t)M_ * E_ * 2);     // v (b,h,d,s)
  half_t* aoh   = (half_t*)alloc((size_t)M_ * E_ * 2);     // attn out (b,s,E)
  half_t* wqT   = (half_t*)alloc((size_t)E_ * DC_ * 2);
  half_t* wqcT  = (half_t*)alloc((size_t)DC_ * NC_ * 2);
  half_t* wqrT  = (half_t*)alloc((size_t)E_ * DR_ * 2);
  half_t* wkvT  = (half_t*)alloc((size_t)E_ * DC_ * 2);
  half_t* wkrT  = (half_t*)alloc((size_t)E_ * DR_ * 2);
  half_t* wkcT  = (half_t*)alloc((size_t)DC_ * NC_ * 2);
  half_t* wvT   = (half_t*)alloc((size_t)DC_ * E_ * 2);
  half_t* woT   = (half_t*)alloc((size_t)E_ * E_ * 2);
  float*  full  = (float*)alloc((size_t)M_ * E_ * 4);      // reused 3x

  const int PT = 256;
  int nElem = M_ * E_;

  // 1) x -> f16
  f32_to_f16_kernel<<<(nElem + PT - 1) / PT, PT, 0, stream>>>(x, xh, nElem);

  // 2) weights -> transposed f16 (N x K row-major)
  auto tr = [&](const float* w, half_t* wt, int din, int dout) {
    int n = din * dout;
    transpose_f16_kernel<<<(n + PT - 1) / PT, PT, 0, stream>>>(w, wt, din, dout);
  };
  tr(wq,  wqT,  E_,  DC_);
  tr(wqc, wqcT, DC_, NC_);
  tr(wqr, wqrT, E_,  DR_);
  tr(wkv, wkvT, E_,  DC_);
  tr(wkr, wkrT, E_,  DR_);
  tr(wkc, wkcT, DC_, NC_);
  tr(wv,  wvT,  DC_, E_);
  tr(wo,  woT,  E_,  E_);

  dim3 blk(128);
  // 3) cq = gelu(x wq + bq), ckv = gelu(x wkv + bkv)   [f16 out]
  gemm_wmma_kernel<true, half_t><<<dim3(M_ / 64, DC_ / 128), blk, 0, stream>>>(
      xh, wqT, bq, cqh, E_, DC_);
  gemm_wmma_kernel<true, half_t><<<dim3(M_ / 64, DC_ / 128), blk, 0, stream>>>(
      xh, wkvT, bkv, ckvh, E_, DC_);

  // 4) q-concat into full (cols 0..895 = cq wqc, cols 896..1023 = x wqr)
  gemm_wmma_kernel<false, float><<<dim3(M_ / 64, NC_ / 128), blk, 0, stream>>>(
      cqh, wqcT, bqc, full, DC_, E_);
  gemm_wmma_kernel<false, float><<<dim3(M_ / 64, DR_ / 128), blk, 0, stream>>>(
      xh, wqrT, bqr, full + NC_, E_, E_);
  rope_gelu_perm_kernel<<<(nElem + PT - 1) / PT, PT, 0, stream>>>(full, qhh);

  // 5) k-concat into full (reused)
  gemm_wmma_kernel<false, float><<<dim3(M_ / 64, NC_ / 128), blk, 0, stream>>>(
      ckvh, wkcT, bkc, full, DC_, E_);
  gemm_wmma_kernel<false, float><<<dim3(M_ / 64, DR_ / 128), blk, 0, stream>>>(
      xh, wkrT, bkr, full + NC_, E_, E_);
  rope_gelu_perm_kernel<<<(nElem + PT - 1) / PT, PT, 0, stream>>>(full, khh);

  // 6) v into full (reused), then permute to (b,h,d,s)
  gemm_wmma_kernel<false, float><<<dim3(M_ / 64, E_ / 128), blk, 0, stream>>>(
      ckvh, wvT, bv, full, DC_, E_);
  v_perm_kernel<<<(nElem + PT - 1) / PT, PT, 0, stream>>>(full, vth);

  // 7) causal flash attention (1 wave per (b,h,32-query tile))
  attn_kernel<<<B_ * H_ * QT2_, 32, 0, stream>>>(qhh, khh, vth, aoh);

  // 8) out = ao wo + bo  [f32 out]
  gemm_wmma_kernel<false, float><<<dim3(M_ / 64, E_ / 128), blk, 0, stream>>>(
      aoh, woT, bo, out, E_, E_);
}
